// VallinaDecoder_24215025614888
// MI455X (gfx1250) — matrix operand order repaired and checked by reading, not compiled, not run on hardware
//
#include <hip/hip_runtime.h>
#include <stdint.h>

#define HD   1024
#define LSEQ 2048
#define BATCH 64
#define KP   1032   // padded LDS row stride in bf16 elements (16B-aligned rows)
#define TPB  256

typedef __attribute__((ext_vector_type(16))) __bf16 v16bf;
typedef __attribute__((ext_vector_type(8)))  float  v8f;

union Frag {
  v16bf    v;
  uint32_t u[8];
  uint4    q[2];
};

__device__ __forceinline__ float leaky(float x) { return x >= 0.f ? x : 0.01f * x; }

__device__ __forceinline__ uint32_t bf16pack(float a, float b) {
  uint32_t ua = __builtin_bit_cast(uint32_t, a);
  uint32_t ub = __builtin_bit_cast(uint32_t, b);
  ua = (ua + 0x7FFFu + ((ua >> 16) & 1u)) >> 16;   // round-to-nearest-even
  ub = (ub + 0x7FFFu + ((ub >> 16) & 1u)) >> 16;
  return (ua & 0xFFFFu) | (ub << 16);
}

// fp32 -> packed bf16 pairs (same row-major layout)
__global__ void cvt_bf16_kernel(const float* __restrict__ src,
                                uint32_t* __restrict__ dst, int npairs) {
  int i = blockIdx.x * TPB + threadIdx.x;
  if (i < npairs) {
    float2 f = ((const float2*)src)[i];
    dst[i] = bf16pack(f.x, f.y);
  }
}

// Stage a 64 x 1024 fp32 panel into LDS as bf16 (row stride KP)
__device__ __forceinline__ void stage_A(const float* __restrict__ A, __bf16* As, int tid) {
  for (int idx = tid; idx < 64 * 512; idx += TPB) {
    int m  = idx >> 9;        // row
    int kp = idx & 511;       // pair index within row
    float2 f = ((const float2*)(A + (size_t)m * HD))[kp];
    ((uint32_t*)(As + m * KP))[kp] = bf16pack(f.x, f.y);
  }
}

// C[64 x N] = A[64 x 1024] @ W^T (+bias, optional leaky). W supplied as bf16 pairs [N][1024].
// 8 waves: wave = (wm 0..3) x (wn 0..1); each wave -> 16 rows x 64 cols (4 WMMA tiles).
__global__ void gemm64_kernel(const float* __restrict__ A,
                              const uint32_t* __restrict__ Wb,   // bf16 pairs, [N][512]
                              const float* __restrict__ bias,
                              float* __restrict__ C, int N, int epi) {
  extern __shared__ char smem[];
  __bf16* As = (__bf16*)smem;
  int tid = threadIdx.x;
  stage_A(A, As, tid);
  __syncthreads();

  int lane = tid & 31, wave = tid >> 5;
  int wm = wave >> 1, wn = wave & 1;
  int nl = lane & 15, half = lane >> 4;
  int nBase = blockIdx.x * 128 + wn * 64;
  int mRow = wm * 16 + nl;
  const uint4* Arow = (const uint4*)(As + (size_t)mRow * KP);

  v8f acc[4] = {};
  for (int kk = 0; kk < HD; kk += 32) {
    Frag a;
    int b16 = (kk + 16 * half) >> 3;         // uint4 index into row
    a.q[0] = Arow[b16];
    a.q[1] = Arow[b16 + 1];
#pragma unroll
    for (int nt = 0; nt < 4; ++nt) {
      int n = nBase + nt * 16 + nl;
      const uint4* Bp = (const uint4*)(Wb + (size_t)n * 512 + ((kk + 16 * half) >> 1));
      Frag b;
      b.q[0] = Bp[0];
      b.q[1] = Bp[1];
      acc[nt] = __builtin_amdgcn_wmma_f32_16x16x32_bf16(false, a.v, false, b.v,
                                                        (short)0, acc[nt], false, false);
    }
  }
  int mBase = wm * 16 + 8 * half;
#pragma unroll
  for (int nt = 0; nt < 4; ++nt) {
    int n = nBase + nt * 16 + nl;
    float bb = bias[n];
#pragma unroll
    for (int r = 0; r < 8; ++r) {
      float vv = acc[nt][r] + bb;
      if (epi) vv = leaky(vv);
      C[(size_t)(mBase + r) * N + n] = vv;
    }
  }
}

// Big attention kernel: 64 (b,l) rows per WG; logit[row] =
//   leaky( sum_a Wl[a]*tanh( leaky((enc@Wv^T)[row,a]+bv[a]) + q[b,a] ) + bl )
__global__ void attn_logits_kernel(const float* __restrict__ enc,     // [B,L,H]
                                   const uint32_t* __restrict__ Wvb,  // bf16 pairs [H][512]
                                   const float* __restrict__ bv,
                                   const float* __restrict__ qadd,    // [B,H] (leaky'd)
                                   const float* __restrict__ Wl,      // [H]
                                   const float* __restrict__ bl,      // [1]
                                   const int*   __restrict__ mask,    // [B,L]
                                   float* __restrict__ logits) {      // [B,L]
  extern __shared__ char smem[];
  __bf16* As  = (__bf16*)smem;                     // 64 x KP bf16
  float* redu = (float*)(smem + 64 * KP * 2);      // 64 partial logits
  int tid = threadIdx.x;
  int rowBase = blockIdx.x * 64;
  int b = rowBase >> 11;                           // 2048 rows per batch, tile-aligned
  stage_A(enc + (size_t)rowBase * HD, As, tid);
  if (tid < 64) redu[tid] = 0.f;
  __syncthreads();

  int lane = tid & 31, wave = tid >> 5;
  int wm = wave >> 1, wn = wave & 1;
  int nl = lane & 15, half = lane >> 4;
  int mRow = wm * 16 + nl;
  const uint4* Arow  = (const uint4*)(As + (size_t)mRow * KP);
  const float* qrow  = qadd + (size_t)b * HD;

  for (int nc = 0; nc < 4; ++nc) {
    int nBase = nc * 256 + wn * 128;
    v8f acc[8] = {};
    for (int kk = 0; kk < HD; kk += 32) {
      Frag a;
      int b16 = (kk + 16 * half) >> 3;
      a.q[0] = Arow[b16];
      a.q[1] = Arow[b16 + 1];
#pragma unroll
      for (int nt = 0; nt < 8; ++nt) {
        int n = nBase + nt * 16 + nl;
        const uint4* Bp = (const uint4*)(Wvb + (size_t)n * 512 + ((kk + 16 * half) >> 1));
        Frag bqf;
        bqf.q[0] = Bp[0];
        bqf.q[1] = Bp[1];
        acc[nt] = __builtin_amdgcn_wmma_f32_16x16x32_bf16(false, a.v, false, bqf.v,
                                                          (short)0, acc[nt], false, false);
      }
    }
    float rowSum[8];
#pragma unroll
    for (int r = 0; r < 8; ++r) rowSum[r] = 0.f;
#pragma unroll
    for (int nt = 0; nt < 8; ++nt) {
      int aIdx  = nBase + nt * 16 + nl;
      float bvv = bv[aIdx];
      float wl  = Wl[aIdx];
      float qv  = qrow[aIdx];
#pragma unroll
      for (int r = 0; r < 8; ++r) {
        float mval = leaky(acc[nt][r] + bvv) + qv;
        rowSum[r] += tanhf(mval) * wl;
      }
    }
#pragma unroll
    for (int r = 0; r < 8; ++r)
      atomicAdd(&redu[wm * 16 + 8 * half + r], rowSum[r]);   // ds_add_f32
  }
  __syncthreads();
  if (tid < 64) {
    int row = rowBase + tid;
    float lg = leaky(redu[tid] + bl[0]);
    logits[row] = mask[row] ? lg : -1e-9f;
  }
}

__global__ void softmax_kernel(const float* __restrict__ logits, float* __restrict__ prob) {
  __shared__ float red[TPB];
  int b = blockIdx.x, tid = threadIdx.x;
  const float* row = logits + (size_t)b * LSEQ;
  float vals[8], mx = -3.4e38f;
#pragma unroll
  for (int i = 0; i < 8; ++i) { vals[i] = row[tid + i * TPB]; mx = fmaxf(mx, vals[i]); }
  red[tid] = mx; __syncthreads();
  for (int s = 128; s > 0; s >>= 1) { if (tid < s) red[tid] = fmaxf(red[tid], red[tid + s]); __syncthreads(); }
  mx = red[0]; __syncthreads();
  float sum = 0.f;
#pragma unroll
  for (int i = 0; i < 8; ++i) { vals[i] = expf(vals[i] - mx); sum += vals[i]; }
  red[tid] = sum; __syncthreads();
  for (int s = 128; s > 0; s >>= 1) { if (tid < s) red[tid] += red[tid + s]; __syncthreads(); }
  float inv = 1.f / red[0];
  float* out = prob + (size_t)b * LSEQ;
#pragma unroll
  for (int i = 0; i < 8; ++i) out[tid + i * TPB] = vals[i] * inv;
}

// attn[b,d] = sum_l prob[b,l] * enc[b,l,d]   (bandwidth-bound)
__global__ void attn_enc_kernel(const float* __restrict__ prob,
                                const float* __restrict__ enc,
                                float* __restrict__ attn) {
  int b = blockIdx.y;
  int d = blockIdx.x * TPB + threadIdx.x;
  const float* p = prob + (size_t)b * LSEQ;
  const float* e = enc + (size_t)b * LSEQ * HD + d;
  float a0 = 0.f, a1 = 0.f, a2 = 0.f, a3 = 0.f;
  for (int l = 0; l < LSEQ; l += 4) {
    a0 += p[l]     * e[(size_t)l * HD];
    a1 += p[l + 1] * e[(size_t)(l + 1) * HD];
    a2 += p[l + 2] * e[(size_t)(l + 2) * HD];
    a3 += p[l + 3] * e[(size_t)(l + 3) * HD];
  }
  attn[(size_t)b * HD + d] = (a0 + a1) + (a2 + a3);
}

// PyTorch GRUCell gating: gates ordered (r, z, n)
__global__ void gru_gate_kernel(const float* __restrict__ gi, const float* __restrict__ gh,
                                const float* __restrict__ hprev, float* __restrict__ hout) {
  int idx = blockIdx.x * TPB + threadIdx.x;     // 0 .. 65535
  int bb = idx >> 10, j = idx & 1023;
  const float* gib = gi + (size_t)bb * 3 * HD;
  const float* ghb = gh + (size_t)bb * 3 * HD;
  float r = 1.f / (1.f + expf(-(gib[j] + ghb[j])));
  float z = 1.f / (1.f + expf(-(gib[HD + j] + ghb[HD + j])));
  float n = tanhf(gib[2 * HD + j] + r * ghb[2 * HD + j]);
  hout[idx] = (1.f - z) * n + z * hprev[idx];
}

__global__ void output_kernel(const float* __restrict__ xo, const float* __restrict__ attn,
                              const float* __restrict__ h2, float* __restrict__ out) {
  int idx = blockIdx.x * TPB + threadIdx.x;
  out[idx] = tanhf(xo[idx] + attn[idx] + h2[idx]);
}

extern "C" void kernel_launch(void* const* d_in, const int* in_sizes, int n_in,
                              void* d_out, int out_size, void* d_ws, size_t ws_size,
                              hipStream_t stream) {
  const float* x      = (const float*)d_in[0];
  const float* hidden = (const float*)d_in[1];
  const float* enc    = (const float*)d_in[2];
  const int*   mask   = (const int*)d_in[3];
  const float* W_ih1  = (const float*)d_in[4];
  const float* W_hh1  = (const float*)d_in[5];
  const float* b_ih1  = (const float*)d_in[6];
  const float* b_hh1  = (const float*)d_in[7];
  const float* W_ih2  = (const float*)d_in[8];
  const float* W_hh2  = (const float*)d_in[9];
  const float* b_ih2  = (const float*)d_in[10];
  const float* b_hh2  = (const float*)d_in[11];
  const float* Wq     = (const float*)d_in[12];
  const float* bq     = (const float*)d_in[13];
  const float* Wv     = (const float*)d_in[14];
  const float* bv     = (const float*)d_in[15];
  const float* Wl     = (const float*)d_in[16];
  const float* bl     = (const float*)d_in[17];
  const float* Wx     = (const float*)d_in[18];
  const float* bx     = (const float*)d_in[19];

  float* out_output = (float*)d_out;                  // [64,1024]
  float* out_h2     = (float*)d_out + BATCH * HD;     // [64,1024]

  char* wsb = (char*)d_ws;
  size_t off = 0;
  auto alloc = [&](size_t bytes) -> void* {
    void* p = wsb + off;
    off += (bytes + 255) & ~(size_t)255;
    return p;
  };
  uint32_t* Wih1b = (uint32_t*)alloc((size_t)3 * HD * HD * 2);
  uint32_t* Whh1b = (uint32_t*)alloc((size_t)3 * HD * HD * 2);
  uint32_t* Wih2b = (uint32_t*)alloc((size_t)3 * HD * HD * 2);
  uint32_t* Whh2b = (uint32_t*)alloc((size_t)3 * HD * HD * 2);
  uint32_t* Wqb   = (uint32_t*)alloc((size_t)HD * HD * 2);
  uint32_t* Wvb   = (uint32_t*)alloc((size_t)HD * HD * 2);
  uint32_t* Wxb   = (uint32_t*)alloc((size_t)HD * HD * 2);
  float* gi     = (float*)alloc((size_t)BATCH * 3 * HD * 4);
  float* gh     = (float*)alloc((size_t)BATCH * 3 * HD * 4);
  float* h1     = (float*)alloc((size_t)BATCH * HD * 4);
  float* qadd   = (float*)alloc((size_t)BATCH * HD * 4);
  float* xo     = (float*)alloc((size_t)BATCH * HD * 4);
  float* logits = (float*)alloc((size_t)BATCH * LSEQ * 4);
  float* prob   = (float*)alloc((size_t)BATCH * LSEQ * 4);
  float* attn   = (float*)alloc((size_t)BATCH * HD * 4);
  (void)ws_size; (void)in_sizes; (void)n_in; (void)out_size;

  int pairs3 = 3 * HD * HD / 2;
  int pairs1 = HD * HD / 2;
  cvt_bf16_kernel<<<(pairs3 + TPB - 1) / TPB, TPB, 0, stream>>>(W_ih1, Wih1b, pairs3);
  cvt_bf16_kernel<<<(pairs3 + TPB - 1) / TPB, TPB, 0, stream>>>(W_hh1, Whh1b, pairs3);
  cvt_bf16_kernel<<<(pairs3 + TPB - 1) / TPB, TPB, 0, stream>>>(W_ih2, Wih2b, pairs3);
  cvt_bf16_kernel<<<(pairs3 + TPB - 1) / TPB, TPB, 0, stream>>>(W_hh2, Whh2b, pairs3);
  cvt_bf16_kernel<<<(pairs1 + TPB - 1) / TPB, TPB, 0, stream>>>(Wq, Wqb, pairs1);
  cvt_bf16_kernel<<<(pairs1 + TPB - 1) / TPB, TPB, 0, stream>>>(Wv, Wvb, pairs1);
  cvt_bf16_kernel<<<(pairs1 + TPB - 1) / TPB, TPB, 0, stream>>>(Wx, Wxb, pairs1);

  size_t shA   = (size_t)64 * KP * 2;                 // 132096 B (of 320 KB/WGP)
  size_t shAtt = shA + 64 * sizeof(float);

  // GRU cell 1
  gemm64_kernel<<<3 * HD / 128, TPB, shA, stream>>>(x, Wih1b, b_ih1, gi, 3 * HD, 0);
  gemm64_kernel<<<3 * HD / 128, TPB, shA, stream>>>(hidden, Whh1b, b_hh1, gh, 3 * HD, 0);
  gru_gate_kernel<<<BATCH * HD / TPB, TPB, 0, stream>>>(gi, gh, hidden, h1);

  // q = leaky(h1@Wq^T+bq), xo = leaky(x@Wx^T+bx)
  gemm64_kernel<<<HD / 128, TPB, shA, stream>>>(h1, Wqb, bq, qadd, HD, 1);
  gemm64_kernel<<<HD / 128, TPB, shA, stream>>>(x, Wxb, bx, xo, HD, 1);

  // attention logits (dominant GEMM, fused epilogue), softmax, weighted sum
  attn_logits_kernel<<<BATCH * LSEQ / 64, TPB, shAtt, stream>>>(enc, Wvb, bv, qadd, Wl, bl,
                                                                mask, logits);
  softmax_kernel<<<BATCH, TPB, 0, stream>>>(logits, prob);
  attn_enc_kernel<<<dim3(HD / TPB, BATCH), TPB, 0, stream>>>(prob, enc, attn);

  // GRU cell 2
  gemm64_kernel<<<3 * HD / 128, TPB, shA, stream>>>(attn, Wih2b, b_ih2, gi, 3 * HD, 0);
  gemm64_kernel<<<3 * HD / 128, TPB, shA, stream>>>(h1, Whh2b, b_hh2, gh, 3 * HD, 0);
  gru_gate_kernel<<<BATCH * HD / TPB, TPB, 0, stream>>>(gi, gh, h1, out_h2);

  // output = tanh(leaky(x@Wx^T+bx) + attn + h2)
  output_kernel<<<BATCH * HD / TPB, TPB, 0, stream>>>(xo, attn, out_h2, out_output);
}
